// MambaEncoder_6339371729243
// MI455X (gfx1250) — compile-verified, matrix operand order checked by reading, not run on hardware
//
#include <hip/hip_runtime.h>
#include <hip/hip_bf16.h>

// ---------------------------------------------------------------------------
// Mamba encoder for MI455X (gfx1250), wave32, WMMA f32<-f16 16x16x32.
// f16 operand staging + 2x2 register-blocked WMMA GEMMs, templated epilogues.
// B=8, L=1024, D_IN=64, D_MODEL=256, D_INNER=512, D_STATE=16, DT_RANK=16.
// ---------------------------------------------------------------------------

typedef __attribute__((ext_vector_type(16))) _Float16 v16h;
typedef __attribute__((ext_vector_type(8)))  _Float16 h8;
typedef __attribute__((ext_vector_type(4)))  _Float16 h4;
typedef __attribute__((ext_vector_type(8)))  float    v8f;

#define ML      8192
#define DMODEL  256
#define DINNER  512
#define DSTATE  16
#define NB      8
#define LSEQ    1024
#define EPSF    1e-5f

__device__ __forceinline__ float siluf(float x) {
    return x * __frcp_rn(1.0f + __expf(-x));
}
__device__ __forceinline__ float softplusf(float x) {
    return (x > 20.0f) ? x : __logf(1.0f + __expf(x));
}

// ---------------------------------------------------------------------------
// f32 -> f16 elementwise convert (4-wide).  n must be a multiple of 4.
// ---------------------------------------------------------------------------
__global__ void cvt_f32_f16(const float* __restrict__ in,
                            _Float16* __restrict__ out, int n)
{
    const int i = (blockIdx.x * blockDim.x + threadIdx.x) * 4;
    if (i >= n) return;
    const float4 v = *(const float4*)(in + i);
    h4 o;
    o[0] = (_Float16)v.x; o[1] = (_Float16)v.y;
    o[2] = (_Float16)v.z; o[3] = (_Float16)v.w;
    *(h4*)(out + i) = o;
}

// ---------------------------------------------------------------------------
// WMMA GEMM, f16 operands, f32 accumulate.
//   C[M,N] = act( A[M,K] * Bt[N,K]^T + bias )  (+= C if RES)
// Template: K (16 or multiple of 32), BM x BN 16-tiles per wave, and
// compile-time epilogue flags so the unrolled store loop is branch-free.
// A fragment = two 16-byte b128 loads (halfs[0..7] at kb, halfs[8..15] at
// kb+16, kb = k0 + hi*8).  K=16 -> upper half compile-time zero.
// ---------------------------------------------------------------------------
template <int K, int BM, int BN, bool HASBIAS, bool SOFTPLUS, bool RES, bool HASC16>
__global__ void wmma_gemm(const _Float16* __restrict__ A, int lda,
                          const _Float16* __restrict__ Bt, int ldb,
                          const float* __restrict__ bias,
                          float* __restrict__ C,
                          _Float16* __restrict__ C16, int ldc,
                          int M, int N)
{
    const int tilesN = N / (16 * BN);
    const int totalTiles = (M / (16 * BM)) * tilesN;
    const int tile = blockIdx.x * (blockDim.x >> 5) + (threadIdx.x >> 5);
    if (tile >= totalTiles) return;

    const int tm = tile / tilesN;
    const int tn = tile % tilesN;

    const int lane = threadIdx.x & 31;
    const int hi   = lane >> 4;
    const int mr   = lane & 15;

    const _Float16* __restrict__ Arow[BM];
    const _Float16* __restrict__ Brow[BN];
#pragma unroll
    for (int i = 0; i < BM; ++i)
        Arow[i] = A + (size_t)((tm * BM + i) * 16 + mr) * lda + hi * 8;
#pragma unroll
    for (int j = 0; j < BN; ++j)
        Brow[j] = Bt + (size_t)((tn * BN + j) * 16 + mr) * ldb + hi * 8;

    v8f acc[BM][BN];
#pragma unroll
    for (int i = 0; i < BM; ++i)
#pragma unroll
        for (int j = 0; j < BN; ++j)
            acc[i][j] = v8f{};

#pragma unroll 4
    for (int k0 = 0; k0 < K; k0 += 32) {
        v16h av[BM], bv[BN];
#pragma unroll
        for (int i = 0; i < BM; ++i) {
            __builtin_prefetch(Arow[i] + k0 + 256, 0, 1);
            const h8 r1 = *(const h8*)(Arow[i] + k0);
            h8 r2;
            if constexpr (K % 32 == 0) r2 = *(const h8*)(Arow[i] + k0 + 16);
            else                       r2 = h8{};
            av[i] = __builtin_shufflevector(r1, r2, 0,1,2,3,4,5,6,7,
                                                    8,9,10,11,12,13,14,15);
        }
#pragma unroll
        for (int j = 0; j < BN; ++j) {
            __builtin_prefetch(Brow[j] + k0 + 256, 0, 1);
            const h8 r1 = *(const h8*)(Brow[j] + k0);
            h8 r2;
            if constexpr (K % 32 == 0) r2 = *(const h8*)(Brow[j] + k0 + 16);
            else                       r2 = h8{};
            bv[j] = __builtin_shufflevector(r1, r2, 0,1,2,3,4,5,6,7,
                                                    8,9,10,11,12,13,14,15);
        }
#pragma unroll
        for (int i = 0; i < BM; ++i)
#pragma unroll
            for (int j = 0; j < BN; ++j)
                acc[i][j] = __builtin_amdgcn_wmma_f32_16x16x32_f16(
                                false, av[i], false, bv[j],
                                (short)0, acc[i][j], false, false);
    }

#pragma unroll
    for (int j = 0; j < BN; ++j) {
        const int n = (tn * BN + j) * 16 + mr;
        float bval = 0.0f;
        if constexpr (HASBIAS) bval = bias[n];
#pragma unroll
        for (int i = 0; i < BM; ++i) {
#pragma unroll
            for (int r = 0; r < 8; ++r) {
                const int m = (tm * BM + i) * 16 + hi * 8 + r;
                float v = acc[i][j][r] + bval;
                if constexpr (SOFTPLUS) v = softplusf(v);
                float* cp = C + (size_t)m * ldc + n;
                if constexpr (RES) v += *cp;
                *cp = v;
                if constexpr (HASC16) C16[(size_t)m * ldc + n] = (_Float16)v;
            }
        }
    }
}

// ---------------------------------------------------------------------------
// LayerNorm over rows of width 256.  One wave32 per row, 8 elems/lane.
// Writes f32 and f16 copies (f16 feeds the WMMA GEMMs).
// ---------------------------------------------------------------------------
__global__ void layernorm_rows(const float* __restrict__ in,
                               const float* __restrict__ g,
                               const float* __restrict__ b,
                               float* __restrict__ out,
                               _Float16* __restrict__ out16, int rows)
{
    const int wid  = (blockIdx.x * blockDim.x + threadIdx.x) >> 5;
    const int lane = threadIdx.x & 31;
    if (wid >= rows) return;

    const float* __restrict__ r = in + (size_t)wid * DMODEL;
    float v[8];
    float s = 0.0f;
#pragma unroll
    for (int i = 0; i < 8; ++i) { v[i] = r[i * 32 + lane]; s += v[i]; }
#pragma unroll
    for (int o = 16; o > 0; o >>= 1) s += __shfl_xor(s, o, 32);
    const float mu = s * (1.0f / DMODEL);

    float vs = 0.0f;
#pragma unroll
    for (int i = 0; i < 8; ++i) { const float d = v[i] - mu; vs += d * d; }
#pragma unroll
    for (int o = 16; o > 0; o >>= 1) vs += __shfl_xor(vs, o, 32);
    const float inv = __frsqrt_rn(vs * (1.0f / DMODEL) + EPSF);

    float*     __restrict__ w   = out   + (size_t)wid * DMODEL;
    _Float16*  __restrict__ w16 = out16 + (size_t)wid * DMODEL;
#pragma unroll
    for (int i = 0; i < 8; ++i) {
        const int c = i * 32 + lane;
        const float y = (v[i] - mu) * inv * g[c] + b[c];
        w[c]   = y;
        w16[c] = (_Float16)y;
    }
}

// ---------------------------------------------------------------------------
// Causal depthwise conv (D_CONV=4) + bias + SiLU.  Writes f32 (for the scan)
// and f16 (for the wx GEMM).  One thread per (b, t, d).
// ---------------------------------------------------------------------------
__global__ void conv_silu(const float* __restrict__ xz,
                          const float* __restrict__ w,
                          const float* __restrict__ bias,
                          float* __restrict__ xc,
                          _Float16* __restrict__ xc16)
{
    const int idx = blockIdx.x * blockDim.x + threadIdx.x;
    if (idx >= NB * LSEQ * DINNER) return;
    const int d = idx & (DINNER - 1);
    const int t = (idx >> 9) & (LSEQ - 1);
    const int b = idx >> 19;

    float acc = bias[d];
#pragma unroll
    for (int j = 0; j < 4; ++j) {
        const int ts = t - 3 + j;
        if (ts >= 0)
            acc += xz[((size_t)(b * LSEQ + ts)) * (2 * DINNER) + d] * w[d * 4 + j];
    }
    const float y = siluf(acc);
    xc[idx]   = y;
    xc16[idx] = (_Float16)y;
}

// ---------------------------------------------------------------------------
// Selective scan (all fp32).  One block per batch, one thread per d.
// 16-state registers; per-step B/C staged through LDS.  Fuses D-skip and
// SiLU(z) gate; writes f16 for the wout GEMM.
// ---------------------------------------------------------------------------
__global__ void selective_scan(const float* __restrict__ xc,
                               const float* __restrict__ dtb,
                               const float* __restrict__ xdbl,
                               const float* __restrict__ xz,
                               const float* __restrict__ A_log,
                               const float* __restrict__ Dp,
                               _Float16* __restrict__ yb16)
{
    const int b = blockIdx.x;
    const int d = threadIdx.x;

    float a[DSTATE];
#pragma unroll
    for (int s = 0; s < DSTATE; ++s) a[s] = -__expf(A_log[d * DSTATE + s]);
    const float Dv = Dp[d];

    float hst[DSTATE];
#pragma unroll
    for (int s = 0; s < DSTATE; ++s) hst[s] = 0.0f;

    __shared__ float bc[2 * DSTATE];
    const size_t rowBase = (size_t)b * LSEQ;

    for (int t = 0; t < LSEQ; ++t) {
        const size_t r = rowBase + t;
        __syncthreads();
        if (threadIdx.x < 2 * DSTATE)
            bc[threadIdx.x] = xdbl[r * 48 + DSTATE + threadIdx.x]; // Bm | Cm
        __syncthreads();

        const float dtv = dtb[r * DINNER + d];
        const float xv  = xc [r * DINNER + d];
        const float dtx = dtv * xv;
        float y = 0.0f;
#pragma unroll
        for (int s = 0; s < DSTATE; ++s) {
            const float dA = __expf(dtv * a[s]);
            hst[s] = dA * hst[s] + dtx * bc[s];
            y += hst[s] * bc[DSTATE + s];
        }
        y += xv * Dv;
        const float zv = xz[r * (2 * DINNER) + DINNER + d];
        yb16[r * DINNER + d] = (_Float16)(y * siluf(zv));
    }
}

// ---------------------------------------------------------------------------
// out[b, c] = mean over t of hn[b, t, c]
// ---------------------------------------------------------------------------
__global__ void mean_over_l(const float* __restrict__ hn, float* __restrict__ out)
{
    const int b = blockIdx.x;
    const int c = threadIdx.x;
    float s = 0.0f;
    for (int t = 0; t < LSEQ; ++t)
        s += hn[((size_t)b * LSEQ + t) * DMODEL + c];
    out[b * DMODEL + c] = s * (1.0f / LSEQ);
}

// ---------------------------------------------------------------------------
// Host orchestration
// ---------------------------------------------------------------------------
template <int K, int BM, int BN, bool HASBIAS, bool SOFTPLUS, bool RES, bool HASC16>
static inline void gemm_launch(hipStream_t stream,
                               const _Float16* A, int lda,
                               const _Float16* Bt, int ldb,
                               const float* bias, float* C, _Float16* C16,
                               int ldc, int M, int N)
{
    const int tiles  = (M / (16 * BM)) * (N / (16 * BN));
    const int blocks = (tiles + 7) / 8;   // 8 waves per block
    wmma_gemm<K, BM, BN, HASBIAS, SOFTPLUS, RES, HASC16>
        <<<blocks, 256, 0, stream>>>(A, lda, Bt, ldb, bias, C, C16, ldc, M, N);
}

static inline void cvt_launch(hipStream_t stream, const float* in,
                              _Float16* out, int n)
{
    cvt_f32_f16<<<(n / 4 + 255) / 256, 256, 0, stream>>>(in, out, n);
}

extern "C" void kernel_launch(void* const* d_in, const int* in_sizes, int n_in,
                              void* d_out, int out_size, void* d_ws, size_t ws_size,
                              hipStream_t stream)
{
    (void)in_sizes; (void)n_in; (void)out_size; (void)ws_size;

    const float* x        = (const float*)d_in[0];   // (8,1024,64)
    const float* inp_w    = (const float*)d_in[1];   // (256,64)
    const float* inp_b    = (const float*)d_in[2];   // (256)
    const float* ln_g     = (const float*)d_in[3];   // (4,256)
    const float* ln_b     = (const float*)d_in[4];   // (4,256)
    const float* win_w    = (const float*)d_in[5];   // (4,1024,256)
    const float* conv_w   = (const float*)d_in[6];   // (4,512,4)
    const float* conv_b   = (const float*)d_in[7];   // (4,512)
    const float* wx_w     = (const float*)d_in[8];   // (4,48,512)
    const float* wdt_w    = (const float*)d_in[9];   // (4,512,16)
    const float* wdt_b    = (const float*)d_in[10];  // (4,512)
    const float* A_log    = (const float*)d_in[11];  // (4,512,16)
    const float* D_p      = (const float*)d_in[12];  // (4,512)
    const float* wout_w   = (const float*)d_in[13];  // (4,256,512)
    const float* out_ln_g = (const float*)d_in[14];  // (256)
    const float* out_ln_b = (const float*)d_in[15];  // (256)

    // ---- fp32 scratch ----
    float* ws   = (float*)d_ws;
    float* h    = ws;                     // 8192*256
    float* hn   = h    + ML * DMODEL;     // 8192*256
    float* xz   = hn   + ML * DMODEL;     // 8192*1024
    float* xc   = xz   + ML * 2 * DINNER; // 8192*512
    float* xdbl = xc   + ML * DINNER;     // 8192*48
    float* dtb  = xdbl + ML * 48;         // 8192*512

    // ---- f16 scratch (after the fp32 region; 16B aligned) ----
    _Float16* x16    = (_Float16*)(dtb + ML * DINNER); // 8192*64
    _Float16* hn16   = x16   + ML * 64;    // 8192*256
    _Float16* xc16   = hn16  + ML * DMODEL;// 8192*512
    _Float16* xdbl16 = xc16  + ML * DINNER;// 8192*48
    _Float16* yb16   = xdbl16+ ML * 48;    // 8192*512
    _Float16* inpw16 = yb16  + ML * DINNER;// 256*64
    _Float16* win16  = inpw16+ DMODEL*64;  // 4*1024*256
    _Float16* wx16   = win16 + 4*2*DINNER*DMODEL; // 4*48*512
    _Float16* wdt16  = wx16  + 4*48*DINNER;       // 4*512*16
    _Float16* wout16 = wdt16 + 4*DINNER*16;       // 4*256*512

    // ---- one-time weight / input conversions ----
    cvt_launch(stream, x,      x16,    ML * 64);
    cvt_launch(stream, inp_w,  inpw16, DMODEL * 64);
    cvt_launch(stream, win_w,  win16,  4 * 2 * DINNER * DMODEL);
    cvt_launch(stream, wx_w,   wx16,   4 * 48 * DINNER);
    cvt_launch(stream, wdt_w,  wdt16,  4 * DINNER * 16);
    cvt_launch(stream, wout_w, wout16, 4 * DMODEL * DINNER);

    // Input projection: h = x @ inp_w^T + inp_b          (K=64, 2x2, bias)
    gemm_launch<64, 2, 2, true, false, false, false>(
        stream, x16, 64, inpw16, 64, inp_b, h, nullptr, DMODEL, ML, DMODEL);

    for (int l = 0; l < 4; ++l) {
        // hn = LayerNorm(h)  (f32 + f16)
        layernorm_rows<<<(ML * 32) / 256, 256, 0, stream>>>(
            h, ln_g + l * DMODEL, ln_b + l * DMODEL, hn, hn16, ML);

        // xz = hn @ win_w[l]^T      (8192 x 1024, K=256, 2x2)
        gemm_launch<256, 2, 2, false, false, false, false>(
            stream, hn16, DMODEL, win16 + (size_t)l * 2 * DINNER * DMODEL,
            DMODEL, nullptr, xz, nullptr, 2 * DINNER, ML, 2 * DINNER);

        // xc = SiLU(causal depthwise conv(x_in) + conv_b)  (f32 + f16)
        conv_silu<<<(NB * LSEQ * DINNER) / 256, 256, 0, stream>>>(
            xz, conv_w + (size_t)l * DINNER * 4, conv_b + l * DINNER, xc, xc16);

        // x_dbl = xc @ wx_w[l]^T    (8192 x 48, K=512, 1x1; f32 + f16 out)
        gemm_launch<512, 1, 1, false, false, false, true>(
            stream, xc16, DINNER, wx16 + (size_t)l * 48 * DINNER, DINNER,
            nullptr, xdbl, xdbl16, 48, ML, 48);

        // dt = softplus(dt_r @ wdt_w[l]^T + wdt_b[l])   (K=16, 2x2)
        gemm_launch<16, 2, 2, true, true, false, false>(
            stream, xdbl16, 48, wdt16 + (size_t)l * DINNER * 16, 16,
            wdt_b + l * DINNER, dtb, nullptr, DINNER, ML, DINNER);

        // y = (scan + xc*D) * SiLU(z)  -> f16
        selective_scan<<<NB, DINNER, 0, stream>>>(
            xc, dtb, xdbl, xz,
            A_log + (size_t)l * DINNER * DSTATE, D_p + l * DINNER, yb16);

        // h += y @ wout_w[l]^T      (K=512, 2x2, residual)
        gemm_launch<512, 2, 2, false, false, true, false>(
            stream, yb16, DINNER, wout16 + (size_t)l * DMODEL * DINNER, DINNER,
            nullptr, h, nullptr, DMODEL, ML, DMODEL);
    }

    // Final LayerNorm + mean over L
    layernorm_rows<<<(ML * 32) / 256, 256, 0, stream>>>(
        h, out_ln_g, out_ln_b, hn, hn16, ML);
    mean_over_l<<<NB, DMODEL, 0, stream>>>(hn, (float*)d_out);
}
